// PotGNN_36069135352228
// MI455X (gfx1250) — compile-verified, matrix operand order checked by reading, not run on hardware
//
#include <hip/hip_runtime.h>
#include <math.h>

typedef __attribute__((ext_vector_type(16))) _Float16 v16h;
typedef __attribute__((ext_vector_type(8)))  _Float16 v8h;
typedef __attribute__((ext_vector_type(8)))  float    v8f;
typedef __attribute__((ext_vector_type(4)))  float    v4f;

#define DN 128
#define DE 128
#define D2 256          // 2*DE
#define D3 640          // 3*DN + 2*DE
#define LN_EPS 1e-5f

__device__ __forceinline__ float sigmoidf_(float x) { return 1.0f / (1.0f + __expf(-x)); }

// Concatenate two 8-half vectors into one WMMA v16h fragment.
__device__ __forceinline__ v16h cat8(v8h lo, v8h hi) {
    return __builtin_shufflevector(lo, hi, 0,1,2,3,4,5,6,7,8,9,10,11,12,13,14,15);
}

// Load one B fragment (16x16x32 f16 layout): lane L holds column (L&15),
// halves K = {kb+off .. kb+off+7, kb+16+off .. kb+16+off+7}, off = (L>=16)*8.
// All addresses are 16B-aligned -> two global_load_b128 per fragment.
__device__ __forceinline__ v16h load_bfrag(const _Float16* __restrict__ wp, int kb, int off) {
    v8h lo = *(const v8h*)(wp + kb + off);
    v8h hi = *(const v8h*)(wp + kb + 16 + off);
    return cat8(lo, hi);
}

// ---------------------------------------------------------------------------
// Convert W2 [256,128] and W3 [256,640] fp32 -> fp16 once per launch.
// ---------------------------------------------------------------------------
__global__ void convert_weights_kernel(const float* __restrict__ W2,
                                       const float* __restrict__ W3,
                                       _Float16* __restrict__ W2h,
                                       _Float16* __restrict__ W3h) {
    int idx = blockIdx.x * blockDim.x + threadIdx.x;
    const int n2 = D2 * DN;   // 32768
    const int n3 = D2 * D3;   // 163840
    if (idx < n2) W2h[idx] = (_Float16)W2[idx];
    if (idx < n3) W3h[idx] = (_Float16)W3[idx];
}

// ---------------------------------------------------------------------------
// c2 branch: one wave per 16 edges, fully register-resident (no LDS).
//   A (16x128 f16)  = node[i[e]] * node[j[e]]
//   GEMM [16,128]x[128,256] -> LN(256) -> sigmoid*tanh gate -> LN(128) -> out
// LN statistics per accumulator slot r (rows r / r+8) via 16-lane-half
// shuffles; the gate pairs C-tile nt with C-tile nt+8 at identical VGPR/lane.
// ---------------------------------------------------------------------------
__global__ __launch_bounds__(32) void c2_kernel(
    const float* __restrict__ node,
    const int*   __restrict__ ei, const int* __restrict__ ej,
    const _Float16* __restrict__ W2h, const float* __restrict__ b2,
    const float* __restrict__ g1, const float* __restrict__ be1,
    const float* __restrict__ g2, const float* __restrict__ be2,
    float* __restrict__ out, int E)
{
    const int lane = threadIdx.x;
    const int nidx = lane & 15;          // A row index / C column index
    const int hi   = lane >> 4;
    const int off  = hi * 8;
    const int e0   = blockIdx.x * 16;
    if (e0 >= E) return;
    const int e    = e0 + nidx;

    // --- A fragments: elementwise product, f32 vector loads -> f16 ---
    const float* pi = node + (size_t)ei[e] * DN;
    const float* pj = node + (size_t)ej[e] * DN;
    v16h a[4];
    #pragma unroll
    for (int ks = 0; ks < 4; ++ks) {
        const int kb = ks * 32;
        v4f x0 = *(const v4f*)(pi + kb + off);
        v4f x1 = *(const v4f*)(pi + kb + off + 4);
        v4f x2 = *(const v4f*)(pi + kb + 16 + off);
        v4f x3 = *(const v4f*)(pi + kb + 16 + off + 4);
        v4f y0 = *(const v4f*)(pj + kb + off);
        v4f y1 = *(const v4f*)(pj + kb + off + 4);
        v4f y2 = *(const v4f*)(pj + kb + 16 + off);
        v4f y3 = *(const v4f*)(pj + kb + 16 + off + 4);
        #pragma unroll
        for (int q = 0; q < 4; ++q) {
            a[ks][q]      = (_Float16)(x0[q] * y0[q]);
            a[ks][4 + q]  = (_Float16)(x1[q] * y1[q]);
            a[ks][8 + q]  = (_Float16)(x2[q] * y2[q]);
            a[ks][12 + q] = (_Float16)(x3[q] * y3[q]);
        }
    }

    // --- GEMM: 16 N-tiles x 4 K-steps, all C tiles stay in registers ---
    v8f c[16];
    #pragma unroll
    for (int nt = 0; nt < 16; ++nt) {
        const int col = nt * 16 + nidx;
        const _Float16* wp = W2h + (size_t)col * DN;
        v8f acc = {};
        #pragma unroll
        for (int ks = 0; ks < 4; ++ks)
            acc = __builtin_amdgcn_wmma_f32_16x16x32_f16(
                    false, a[ks], false, load_bfrag(wp, ks * 32, off),
                    (short)0, acc, false, false);
        const float bias = b2[col];
        #pragma unroll
        for (int r = 0; r < 8; ++r) acc[r] += bias;
        c[nt] = acc;
    }

    // --- LN(256): per-slot row sums, reduced across each 16-lane half ---
    v8f s = {}, sq = {};
    #pragma unroll
    for (int nt = 0; nt < 16; ++nt)
        #pragma unroll
        for (int r = 0; r < 8; ++r) { s[r] += c[nt][r]; sq[r] += c[nt][r] * c[nt][r]; }
    v8f mean, rinv;
    #pragma unroll
    for (int r = 0; r < 8; ++r) {
        float sr = s[r], qr = sq[r];
        #pragma unroll
        for (int m = 1; m < 16; m <<= 1) { sr += __shfl_xor(sr, m, 16); qr += __shfl_xor(qr, m, 16); }
        mean[r] = sr * (1.0f / D2);
        rinv[r] = rsqrtf(qr * (1.0f / D2) - mean[r] * mean[r] + LN_EPS);
    }

    // --- gate: tile nt (filter) with tile nt+8 (core), register-local ---
    v8f act[8];
    v8f s2 = {}, sq2 = {};
    #pragma unroll
    for (int nt = 0; nt < 8; ++nt) {
        const int cf = nt * 16 + nidx;
        const float gf = g1[cf], bf = be1[cf];
        const float gg = g1[cf + 128], bg = be1[cf + 128];
        #pragma unroll
        for (int r = 0; r < 8; ++r) {
            const float f = (c[nt][r]     - mean[r]) * rinv[r] * gf + bf;
            const float g = (c[nt + 8][r] - mean[r]) * rinv[r] * gg + bg;
            const float v = sigmoidf_(f) * tanhf(g);
            act[nt][r] = v; s2[r] += v; sq2[r] += v * v;
        }
    }

    // --- LN(128) of gated activations ---
    v8f mean2, rinv2;
    #pragma unroll
    for (int r = 0; r < 8; ++r) {
        float sr = s2[r], qr = sq2[r];
        #pragma unroll
        for (int m = 1; m < 16; m <<= 1) { sr += __shfl_xor(sr, m, 16); qr += __shfl_xor(qr, m, 16); }
        mean2[r] = sr * (1.0f / DE);
        rinv2[r] = rsqrtf(qr * (1.0f / DE) - mean2[r] * mean2[r] + LN_EPS);
    }

    #pragma unroll
    for (int nt = 0; nt < 8; ++nt) {
        const int cf = nt * 16 + nidx;
        const float gv = g2[cf], bv = be2[cf];
        #pragma unroll
        for (int r = 0; r < 8; ++r)
            out[(size_t)(e0 + r + hi * 8) * DE + cf] =
                (act[nt][r] - mean2[r]) * rinv2[r] * gv + bv;
    }
}

// ---------------------------------------------------------------------------
// c3 branch: one wave per 16 triplets, register-resident.
//   A (16x640 f16) = concat of 5 gathered 128-vectors
//   GEMM [16,640]x[640,256] -> LN(256) -> gate -> atomic scatter by index_ji
// ---------------------------------------------------------------------------
__global__ __launch_bounds__(32) void c3_kernel(
    const float* __restrict__ node, const float* __restrict__ edge,
    const int* __restrict__ idx_i, const int* __restrict__ idx_j,
    const int* __restrict__ idx_k, const int* __restrict__ idx_ji,
    const int* __restrict__ idx_kj,
    const _Float16* __restrict__ W3h, const float* __restrict__ b3,
    const float* __restrict__ g1, const float* __restrict__ be1,
    float* __restrict__ c3_sum, int T)
{
    const int lane = threadIdx.x;
    const int nidx = lane & 15;
    const int hi   = lane >> 4;
    const int off  = hi * 8;
    const int t0   = blockIdx.x * 16;
    if (t0 >= T) return;
    const int t    = t0 + nidx;

    const float* seg[5];
    seg[0] = node + (size_t)idx_i[t]  * DN;
    seg[1] = node + (size_t)idx_j[t]  * DN;
    seg[2] = node + (size_t)idx_k[t]  * DN;
    seg[3] = edge + (size_t)idx_ji[t] * DE;
    seg[4] = edge + (size_t)idx_kj[t] * DE;

    v16h a[20];
    #pragma unroll
    for (int ks = 0; ks < 20; ++ks) {
        const float* p = seg[ks >> 2];
        const int kb = (ks & 3) * 32;
        v4f x0 = *(const v4f*)(p + kb + off);
        v4f x1 = *(const v4f*)(p + kb + off + 4);
        v4f x2 = *(const v4f*)(p + kb + 16 + off);
        v4f x3 = *(const v4f*)(p + kb + 16 + off + 4);
        #pragma unroll
        for (int q = 0; q < 4; ++q) {
            a[ks][q]      = (_Float16)x0[q];
            a[ks][4 + q]  = (_Float16)x1[q];
            a[ks][8 + q]  = (_Float16)x2[q];
            a[ks][12 + q] = (_Float16)x3[q];
        }
    }

    v8f c[16];
    #pragma unroll
    for (int nt = 0; nt < 16; ++nt) {
        const int col = nt * 16 + nidx;
        const _Float16* wp = W3h + (size_t)col * D3;
        v8f acc = {};
        #pragma unroll
        for (int ks = 0; ks < 20; ++ks)
            acc = __builtin_amdgcn_wmma_f32_16x16x32_f16(
                    false, a[ks], false, load_bfrag(wp, ks * 32, off),
                    (short)0, acc, false, false);
        const float bias = b3[col];
        #pragma unroll
        for (int r = 0; r < 8; ++r) acc[r] += bias;
        c[nt] = acc;
    }

    // --- LN(256) stats in registers ---
    v8f s = {}, sq = {};
    #pragma unroll
    for (int nt = 0; nt < 16; ++nt)
        #pragma unroll
        for (int r = 0; r < 8; ++r) { s[r] += c[nt][r]; sq[r] += c[nt][r] * c[nt][r]; }
    v8f mean, rinv;
    #pragma unroll
    for (int r = 0; r < 8; ++r) {
        float sr = s[r], qr = sq[r];
        #pragma unroll
        for (int m = 1; m < 16; m <<= 1) { sr += __shfl_xor(sr, m, 16); qr += __shfl_xor(qr, m, 16); }
        mean[r] = sr * (1.0f / D2);
        rinv[r] = rsqrtf(qr * (1.0f / D2) - mean[r] * mean[r] + LN_EPS);
    }

    // --- destination rows for scatter (row r in slot order r + hi*8) ---
    float* dpt[8];
    #pragma unroll
    for (int r = 0; r < 8; ++r)
        dpt[r] = c3_sum + (size_t)idx_ji[t0 + r + hi * 8] * DE;

    // --- gate + atomic scatter-add ---
    #pragma unroll
    for (int nt = 0; nt < 8; ++nt) {
        const int cf = nt * 16 + nidx;
        const float gf = g1[cf], bf = be1[cf];
        const float gg = g1[cf + 128], bg = be1[cf + 128];
        #pragma unroll
        for (int r = 0; r < 8; ++r) {
            const float f = (c[nt][r]     - mean[r]) * rinv[r] * gf + bf;
            const float g = (c[nt + 8][r] - mean[r]) * rinv[r] * gg + bg;
            atomicAdd(dpt[r] + cf, sigmoidf_(f) * tanhf(g));
        }
    }
}

// ---------------------------------------------------------------------------
// finalize: out = tanh(edge + c2_emb(out) + LN(c3_sum)); one wave -> 4 rows.
// ---------------------------------------------------------------------------
__global__ __launch_bounds__(256) void finalize_kernel(
    const float* __restrict__ edge, const float* __restrict__ c3_sum,
    const float* __restrict__ g2, const float* __restrict__ be2,
    float* __restrict__ out, int E)
{
    const int wave = threadIdx.x >> 5;
    const int lane = threadIdx.x & 31;
    #pragma unroll
    for (int rr = 0; rr < 4; ++rr) {
        const int e = blockIdx.x * 32 + wave * 4 + rr;
        if (e >= E) continue;
        const float* cp = c3_sum + (size_t)e * DE;
        const v4f v = *(const v4f*)(cp + lane * 4);
        float s  = v[0] + v[1] + v[2] + v[3];
        float sq = v[0]*v[0] + v[1]*v[1] + v[2]*v[2] + v[3]*v[3];
        #pragma unroll
        for (int m = 1; m < 32; m <<= 1) {
            s  += __shfl_xor(s,  m, 32);
            sq += __shfl_xor(sq, m, 32);
        }
        const float mean = s * (1.0f / DE);
        const float rinv = rsqrtf(sq * (1.0f / DE) - mean * mean + LN_EPS);
        const float* ep = edge + (size_t)e * DE;
        float*       op = out  + (size_t)e * DE;
        #pragma unroll
        for (int q = 0; q < 4; ++q) {
            const int cidx = lane * 4 + q;
            const float c3n = (v[q] - mean) * rinv * g2[cidx] + be2[cidx];
            op[cidx] = tanhf(ep[cidx] + op[cidx] + c3n);
        }
    }
}

// ---------------------------------------------------------------------------
extern "C" void kernel_launch(void* const* d_in, const int* in_sizes, int n_in,
                              void* d_out, int out_size, void* d_ws, size_t ws_size,
                              hipStream_t stream) {
    const float* node  = (const float*)d_in[0];
    const float* edge  = (const float*)d_in[1];
    const int*   ei    = (const int*)d_in[2];
    const int*   ej    = (const int*)d_in[3];
    const int*   idx_i = (const int*)d_in[4];
    const int*   idx_j = (const int*)d_in[5];
    const int*   idx_k = (const int*)d_in[6];
    const int*   idx_ji= (const int*)d_in[7];
    const int*   idx_kj= (const int*)d_in[8];
    const float* W2    = (const float*)d_in[9];
    const float* b2    = (const float*)d_in[10];
    const float* W3    = (const float*)d_in[11];
    const float* b3    = (const float*)d_in[12];
    const float* g_c2  = (const float*)d_in[13];
    const float* be_c2 = (const float*)d_in[14];
    const float* g_c3  = (const float*)d_in[15];
    const float* be_c3 = (const float*)d_in[16];
    const float* g_c2_2  = (const float*)d_in[17];
    const float* be_c2_2 = (const float*)d_in[18];
    const float* g_c3_2  = (const float*)d_in[19];
    const float* be_c3_2 = (const float*)d_in[20];

    const int E = in_sizes[1] / DE;     // edge_embedding [E,128]
    const int T = in_sizes[4];          // index_i [T]
    float* out  = (float*)d_out;

    // workspace layout: [c3_sum: E*128 f32][W2h: 256*128 f16][W3h: 256*640 f16]
    float*    c3_sum = (float*)d_ws;
    _Float16* W2h    = (_Float16*)((char*)d_ws + (size_t)E * DE * sizeof(float));
    _Float16* W3h    = W2h + (size_t)D2 * DN;
    (void)ws_size; (void)n_in; (void)out_size;

    hipMemsetAsync(c3_sum, 0, (size_t)E * DE * sizeof(float), stream);

    convert_weights_kernel<<<(D2 * D3 + 255) / 256, 256, 0, stream>>>(W2, W3, W2h, W3h);

    c2_kernel<<<(E + 15) / 16, 32, 0, stream>>>(
        node, ei, ej, W2h, b2, g_c2, be_c2, g_c2_2, be_c2_2, out, E);

    c3_kernel<<<(T + 15) / 16, 32, 0, stream>>>(
        node, edge, idx_i, idx_j, idx_k, idx_ji, idx_kj,
        W3h, b3, g_c3, be_c3, c3_sum, T);

    finalize_kernel<<<(E + 31) / 32, 256, 0, stream>>>(
        edge, c3_sum, g_c3_2, be_c3_2, out, E);
}
